// WiredVoltron_42597485642505
// MI455X (gfx1250) — matrix operand order repaired and checked by reading, not compiled, not run on hardware
//
#include <hip/hip_runtime.h>
#include <math.h>

typedef __attribute__((ext_vector_type(16))) _Float16 v16h;
typedef __attribute__((ext_vector_type(8)))  _Float16 v8h;
typedef __attribute__((ext_vector_type(8)))  float    v8f;

#define LUT_N (1 << 17)   // 256 A * 256 operand * 2 C_in distinct network inputs

// is_alu per opcode. NOTE: reference builds routing table from np.zeros, so every
// opcode NOT assigned to a non-ALU group defaults to unit 0 == ALU. Mask bit=1 -> ALU.
__device__ __constant__ unsigned ALU_MASK[8] = {
  0x9C9C989Du, 0x9C9C989Du, 0x9C9C989Du, 0xBEBEBABFu,
  0xD88C8B8Du, 0x888C8888u, 0xFEFEFFFFu, 0xFEFEFBFFu
};

// ---------------- prep kernel 1: fold scalar encoders into 256x32 f16 tables ----
__global__ void enc_table_kernel(const float* __restrict__ W1a, const float* __restrict__ b1a,
                                 const float* __restrict__ W2a, const float* __restrict__ b2a,
                                 const float* __restrict__ W1o, const float* __restrict__ b1o,
                                 const float* __restrict__ W2o, const float* __restrict__ b2o,
                                 _Float16* __restrict__ tabA, _Float16* __restrict__ tabO) {
  int t = blockIdx.x * blockDim.x + threadIdx.x;
  if (t >= 512) return;
  int v = t & 255;
  const float *W1, *b1, *W2, *b2; _Float16* dst;
  if (t < 256) { W1 = W1a; b1 = b1a; W2 = W2a; b2 = b2a; dst = tabA + v * 32; }
  else         { W1 = W1o; b1 = b1o; W2 = W2o; b2 = b2o; dst = tabO + v * 32; }
  float x = (float)v * (1.0f / 255.0f);
  float h[64];
  #pragma unroll
  for (int j = 0; j < 64; ++j) h[j] = fmaxf(fmaf(W1[j], x, b1[j]), 0.0f);
  for (int i = 0; i < 32; ++i) {
    float z = b2[i];
    #pragma unroll
    for (int j = 0; j < 64; ++j) z = fmaf(h[j], W2[j * 32 + i], z);
    dst[i] = (_Float16)(1.0f / (1.0f + __expf(-z)));
  }
}

// ---------------- prep kernel 2: repack weights f32 [K][N] -> f16 N-major [Npad][Kpad]
__global__ void wt_pack_kernel(const float* __restrict__ src, _Float16* __restrict__ dst,
                               int K, int N, int Kpad, int Npad) {
  int i = blockIdx.x * blockDim.x + threadIdx.x;
  if (i >= Kpad * Npad) return;
  int n = i / Kpad, k = i - n * Kpad;
  float v = (k < K && n < N) ? src[k * N + n] : 0.0f;
  dst[i] = (_Float16)v;
}

// ---------------- WMMA fragment loaders (contiguous k = 16*(lane>>4)+s, same in A and B) ----------
// A (16x32 f16): lanes0-15 row M=lane, K=0..15; lanes16-31 K=16..31
__device__ __forceinline__ v16h load_a_frag(const _Float16* buf, int S, int kstep, int lane) {
  int row = lane & 15, grp = lane >> 4;
  const _Float16* p = buf + row * S + kstep * 32 + grp * 16;
  v8h lo = *(const v8h*)p;
  v8h hi = *(const v8h*)(p + 8);
  v16h a;
  #pragma unroll
  for (int i = 0; i < 8; ++i) { a[i] = lo[i]; a[i + 8] = hi[i]; }
  return a;
}
// B (32x16 f16): lane holds column N=lane&15; lanes0-15 K=0..15, lanes16-31 K=16..31.
// Weights stored N-major so this is 16 contiguous halfs (32B aligned).
__device__ __forceinline__ v16h load_b_frag(const _Float16* Wt, int Kpad, int nt, int kstep, int lane) {
  int n = nt * 16 + (lane & 15);
  int kb = kstep * 32 + (lane >> 4) * 16;
  return *(const v16h*)(Wt + n * Kpad + kb);
}

// One dense layer: LDS f16 activations (16 x Kpad) x Wt -> LDS f16 (16 x Ntiles*16)
__device__ __forceinline__ void wmma_layer(const _Float16* in, int Sin, int Ksteps,
                                           const _Float16* __restrict__ Wt, int Kpad,
                                           const float* __restrict__ bias,
                                           _Float16* out, int Sout, int Ntiles,
                                           int lane, bool hard) {
  int nl = lane & 15, grp = lane >> 4;
  for (int nt = 0; nt < Ntiles; ++nt) {
    v8f c = {};
    for (int k = 0; k < Ksteps; ++k) {
      v16h a = load_a_frag(in, Sin, k, lane);
      v16h b = load_b_frag(Wt, Kpad, nt, k, lane);
      c = __builtin_amdgcn_wmma_f32_16x16x32_f16(false, a, false, b, (short)0, c, false, false);
    }
    float bv = bias[nt * 16 + nl];
    #pragma unroll
    for (int v = 0; v < 8; ++v) {  // D layout: M = v + 8*grp, N = nl
      float f = c[v] + bv;
      f = hard ? (f > 0.0f ? 1.0f : 0.0f) : fmaxf(f, 0.0f);
      out[(v + grp * 8) * Sout + nt * 16 + nl] = (_Float16)f;
    }
  }
  __syncthreads();
}

__device__ __forceinline__ float lds_dot(const _Float16* in, int S, int K,
                                         const float* __restrict__ w, float b, int row) {
  float z = b;
  #pragma unroll 8
  for (int j = 0; j < K; ++j) z = fmaf((float)in[row * S + j], w[j], z);
  return z;
}

struct NetPtrs {
  const _Float16 *encA, *encOp;
  const _Float16 *rW1, *rW2, *rW3, *rW4, *cW1, *cW2, *oW1, *oW2, *oW3, *dW1;
  const float *rb1, *rb2, *rb3, *rb4;
  const float *cb1, *cb2, *cw3, *cb3;
  const float *ob1, *ob2, *ob3, *ow4, *ob4;
  const float *db1, *dw2, *db2;
  float *lutA, *lutC, *lutV;
};

// ---------------- main kernel: evaluate all 131072 unique inputs with WMMA ------
__global__ __launch_bounds__(64) void lut_build_kernel(NetPtrs p) {
  __shared__ _Float16 sm_x[2 * 16 * 96];    // x = [encA(32) | encOp(32) | C | 0-pad] -> K=96
  __shared__ _Float16 sm_a[2 * 16 * 256];
  __shared__ _Float16 sm_b[2 * 16 * 128];
  int wave = threadIdx.x >> 5, lane = threadIdx.x & 31;
  _Float16* xb = sm_x + wave * 16 * 96;
  _Float16* bA = sm_a + wave * 16 * 256;
  _Float16* bB = sm_b + wave * 16 * 128;
  int tileBase = (blockIdx.x * 2 + wave) * 16;

  { // build x rows from folded encoder tables; r = (A<<9)|(op<<1)|C
    int row = lane & 15, grp = lane >> 4;
    int r = tileBase + row;
    if (grp == 0) {
      const v8h* src = (const v8h*)(p.encA + (r >> 9) * 32);
      v8h* dst = (v8h*)(xb + row * 96);
      dst[0] = src[0]; dst[1] = src[1]; dst[2] = src[2]; dst[3] = src[3];
      xb[row * 96 + 64] = (_Float16)(float)(r & 1);
      for (int j = 65; j < 80; ++j) xb[row * 96 + j] = (_Float16)0.0f;
    } else {
      const v8h* src = (const v8h*)(p.encOp + ((r >> 1) & 255) * 32);
      v8h* dst = (v8h*)(xb + row * 96 + 32);
      dst[0] = src[0]; dst[1] = src[1]; dst[2] = src[2]; dst[3] = src[3];
      for (int j = 80; j < 96; ++j) xb[row * 96 + j] = (_Float16)0.0f;
    }
  }
  __syncthreads();

  int row = lane & 15;
  // result head: 65->256->128->64->32, hard threshold on final pre-activation
  wmma_layer(xb, 96, 3, p.rW1, 96, p.rb1, bA, 256, 16, lane, false);
  wmma_layer(bA, 256, 8, p.rW2, 256, p.rb2, bB, 128, 8, lane, false);
  wmma_layer(bB, 128, 4, p.rW3, 128, p.rb3, bA, 64, 4, lane, false);
  wmma_layer(bA, 64, 2, p.rW4, 64, p.rb4, bB, 32, 2, lane, true);   // result_hard
  // decoder: 32->64 (WMMA) -> 1 (VALU dot), sigmoid*255
  wmma_layer(bB, 32, 1, p.dW1, 32, p.db1, bA, 64, 4, lane, false);
  {
    float z = lds_dot(bA, 64, 64, p.dw2, p.db2[0], row);
    float s = 1.0f / (1.0f + __expf(-z));
    if (lane < 16) p.lutA[tileBase + row] = fminf(fmaxf(255.0f * s, 0.0f), 255.0f);
  }
  // carry head: 65->128->64->1, sigmoid>0.5 == z>0
  wmma_layer(xb, 96, 3, p.cW1, 96, p.cb1, bA, 128, 8, lane, false);
  wmma_layer(bA, 128, 4, p.cW2, 128, p.cb2, bB, 64, 4, lane, false);
  {
    float z = lds_dot(bB, 64, 64, p.cw3, p.cb3[0], row);
    if (lane < 16) p.lutC[tileBase + row] = (z > 0.0f) ? 1.0f : 0.0f;
  }
  // overflow head: 65->128->64->32->1
  wmma_layer(xb, 96, 3, p.oW1, 96, p.ob1, bA, 128, 8, lane, false);
  wmma_layer(bA, 128, 4, p.oW2, 128, p.ob2, bB, 64, 4, lane, false);
  wmma_layer(bB, 64, 2, p.oW3, 64, p.ob3, bA, 32, 2, lane, false);
  {
    float z = lds_dot(bA, 32, 32, p.ow4, p.ob4[0], row);
    if (lane < 16) p.lutV[tileBase + row] = (z > 0.0f) ? 1.0f : 0.0f;
  }
}

// ---------------- gather: per-sample LUT lookup + routing blend ------------------
__global__ void gather_kernel(const int* __restrict__ A, const int* __restrict__ op,
                              const int* __restrict__ C, const int* __restrict__ opc,
                              const float* __restrict__ lutA, const float* __restrict__ lutC,
                              const float* __restrict__ lutV, float* __restrict__ out, int B) {
  int i = blockIdx.x * blockDim.x + threadIdx.x;
  if (i >= B) return;
  int a = A[i], k = opc[i];
  int r = (a << 9) | (op[i] << 1) | C[i];
  bool alu = (ALU_MASK[(k >> 5) & 7] >> (k & 31)) & 1u;
  out[i]         = alu ? lutA[r] : (float)a;
  out[B + i]     = lutC[r];
  out[2 * B + i] = lutV[r];
}

extern "C" void kernel_launch(void* const* d_in, const int* in_sizes, int n_in,
                              void* d_out, int out_size, void* d_ws, size_t ws_size,
                              hipStream_t stream) {
  (void)n_in; (void)out_size; (void)ws_size;
  const int* A   = (const int*)d_in[0];
  const int* op  = (const int*)d_in[1];
  const int* Cin = (const int*)d_in[2];
  const int* opc = (const int*)d_in[3];
  int B = in_sizes[0];

  // workspace carve-up (256B-aligned chunks)
  char* ws = (char*)d_ws;
  size_t off = 0;
  auto take = [&](size_t bytes) -> char* {
    char* q = ws + off; off += (bytes + 255) & ~(size_t)255; return q;
  };
  _Float16* encA = (_Float16*)take(256 * 32 * 2);
  _Float16* encO = (_Float16*)take(256 * 32 * 2);
  _Float16* rW1 = (_Float16*)take(256 * 96 * 2);
  _Float16* rW2 = (_Float16*)take(128 * 256 * 2);
  _Float16* rW3 = (_Float16*)take(64 * 128 * 2);
  _Float16* rW4 = (_Float16*)take(32 * 64 * 2);
  _Float16* cW1 = (_Float16*)take(128 * 96 * 2);
  _Float16* cW2 = (_Float16*)take(64 * 128 * 2);
  _Float16* oW1 = (_Float16*)take(128 * 96 * 2);
  _Float16* oW2 = (_Float16*)take(64 * 128 * 2);
  _Float16* oW3 = (_Float16*)take(32 * 64 * 2);
  _Float16* dW1 = (_Float16*)take(64 * 32 * 2);
  float* lutA = (float*)take(LUT_N * 4);
  float* lutC = (float*)take(LUT_N * 4);
  float* lutV = (float*)take(LUT_N * 4);

  // fold the scalar encoders
  enc_table_kernel<<<2, 256, 0, stream>>>(
      (const float*)d_in[4], (const float*)d_in[5], (const float*)d_in[6], (const float*)d_in[7],
      (const float*)d_in[8], (const float*)d_in[9], (const float*)d_in[10], (const float*)d_in[11],
      encA, encO);

  // repack weights to f16 N-major [Npad][Kpad]
  struct Job { const float* src; _Float16* dst; int K, N, Kpad, Npad; };
  Job jobs[10] = {
    { (const float*)d_in[16], rW1,  65, 256,  96, 256 },
    { (const float*)d_in[18], rW2, 256, 128, 256, 128 },
    { (const float*)d_in[20], rW3, 128,  64, 128,  64 },
    { (const float*)d_in[22], rW4,  64,  32,  64,  32 },
    { (const float*)d_in[24], cW1,  65, 128,  96, 128 },
    { (const float*)d_in[26], cW2, 128,  64, 128,  64 },
    { (const float*)d_in[30], oW1,  65, 128,  96, 128 },
    { (const float*)d_in[32], oW2, 128,  64, 128,  64 },
    { (const float*)d_in[34], oW3,  64,  32,  64,  32 },
    { (const float*)d_in[12], dW1,  32,  64,  32,  64 },
  };
  for (int j = 0; j < 10; ++j) {
    int tot = jobs[j].Kpad * jobs[j].Npad;
    wt_pack_kernel<<<(tot + 255) / 256, 256, 0, stream>>>(
        jobs[j].src, jobs[j].dst, jobs[j].K, jobs[j].N, jobs[j].Kpad, jobs[j].Npad);
  }

  NetPtrs p;
  p.encA = encA; p.encOp = encO;
  p.rW1 = rW1; p.rW2 = rW2; p.rW3 = rW3; p.rW4 = rW4;
  p.cW1 = cW1; p.cW2 = cW2; p.oW1 = oW1; p.oW2 = oW2; p.oW3 = oW3; p.dW1 = dW1;
  p.rb1 = (const float*)d_in[17]; p.rb2 = (const float*)d_in[19];
  p.rb3 = (const float*)d_in[21]; p.rb4 = (const float*)d_in[23];
  p.cb1 = (const float*)d_in[25]; p.cb2 = (const float*)d_in[27];
  p.cw3 = (const float*)d_in[28]; p.cb3 = (const float*)d_in[29];
  p.ob1 = (const float*)d_in[31]; p.ob2 = (const float*)d_in[33];
  p.ob3 = (const float*)d_in[35]; p.ow4 = (const float*)d_in[36]; p.ob4 = (const float*)d_in[37];
  p.db1 = (const float*)d_in[13]; p.dw2 = (const float*)d_in[14]; p.db2 = (const float*)d_in[15];
  p.lutA = lutA; p.lutC = lutC; p.lutV = lutV;

  // 131072 rows, 16 rows/wave, 2 waves/block
  lut_build_kernel<<<LUT_N / 32, 64, 0, stream>>>(p);

  gather_kernel<<<(B + 255) / 256, 256, 0, stream>>>(
      A, op, Cin, opc, lutA, lutC, lutV, (float*)d_out, B);
}
